// SingleMambaModel_65730179498081
// MI455X (gfx1250) — compile-verified, matrix operand order checked by reading, not run on hardware
//
#include <hip/hip_runtime.h>
#include <hip/hip_bf16.h>
#include <math.h>

// ---- Problem constants (from reference) ----
#define D_MODEL   1024
#define D_STATE   128
#define D_CONV    4
#define HEADDIM   64
#define D_INNER   2048           // EXPAND * D_MODEL
#define NHEADS    32             // D_INNER / HEADDIM
#define CONV_DIM  2304           // D_INNER + 2*D_STATE
#define D_IN_PROJ 4384           // 2*D_INNER + 2*D_STATE + NHEADS
#define BATCH     2
#define SEQLEN    2048
#define MROWS     (BATCH * SEQLEN)   // 4096

typedef __attribute__((ext_vector_type(16))) __bf16 v16bf;
typedef __attribute__((ext_vector_type(8)))  __bf16 v8bf;
typedef __attribute__((ext_vector_type(8)))  float  v8f;

#if defined(__has_builtin)
#if __has_builtin(__builtin_amdgcn_global_load_async_to_lds_b32)
#define HAVE_ASYNC_LDS 1
#endif
#endif

typedef __attribute__((address_space(1))) int* gptr_i32;
typedef __attribute__((address_space(3))) int* lptr_i32;

// -----------------------------------------------------------------------------
// Bulk f32 -> bf16 conversion (one pass; GEMMs then stream pure bf16).
// -----------------------------------------------------------------------------
__global__ __launch_bounds__(256)
void f32_to_bf16_kernel(const float* __restrict__ in, v8bf* __restrict__ out,
                        int n8) {
    int i = blockIdx.x * blockDim.x + threadIdx.x;
    if (i >= n8) return;
    const float* p = in + (size_t)i * 8;
    v8bf v;
#pragma unroll
    for (int j = 0; j < 8; ++j) v[j] = (__bf16)p[j];
    out[i] = v;
}

// -----------------------------------------------------------------------------
// NT GEMM: C[M,N] = A[M,K] @ W[N,K]^T (+ resid).  bf16 operands, f32 accum.
// Block = 128 threads = 4 waves.  Each wave owns a 64Mx32N tile: 4 A-frags x
// 2 B-frags -> 8 WMMAs per K-step at 1.5 b128 loads/WMMA.  Block covers
// 256 rows x 32 cols; all shapes divide exactly (M/256=16, N/32 = 137 or 32).
// -----------------------------------------------------------------------------
__global__ __launch_bounds__(128)
void wmma_gemm_nt(const __bf16* __restrict__ A,    // M x K bf16 row-major
                  const __bf16* __restrict__ W,    // N x K bf16 row-major
                  const float* __restrict__ resid, // M x N or nullptr
                  float* __restrict__ C,           // M x N
                  int M, int N, int K) {
    const int lane = threadIdx.x & 31;
    const int wave = threadIdx.x >> 5;
    const int half = lane >> 4;
    const int r    = lane & 15;

    const int m0 = blockIdx.y * 256 + wave * 64;
    const int n0 = blockIdx.x * 32;

    v8f acc[4][2];
#pragma unroll
    for (int i = 0; i < 4; ++i)
#pragma unroll
        for (int j = 0; j < 2; ++j) acc[i][j] = (v8f){};

    const __bf16* arow[4];
#pragma unroll
    for (int i = 0; i < 4; ++i)
        arow[i] = A + (size_t)(m0 + i * 16 + r) * K + half * 16;
    const __bf16* brow[2];
#pragma unroll
    for (int j = 0; j < 2; ++j)
        brow[j] = W + (size_t)(n0 + j * 16 + r) * K + half * 16;

    for (int k0 = 0; k0 < K; k0 += 32) {
        if (k0 + 32 < K) {  // global_prefetch_b8 for the next K tile
#pragma unroll
            for (int i = 0; i < 4; ++i) __builtin_prefetch(arow[i] + k0 + 32, 0, 1);
#pragma unroll
            for (int j = 0; j < 2; ++j) __builtin_prefetch(brow[j] + k0 + 32, 0, 1);
        }
        v16bf a[4], b[2];
#pragma unroll
        for (int i = 0; i < 4; ++i) a[i] = *(const v16bf*)(arow[i] + k0);
#pragma unroll
        for (int j = 0; j < 2; ++j) b[j] = *(const v16bf*)(brow[j] + k0);
#pragma unroll
        for (int i = 0; i < 4; ++i) {
#pragma unroll
            for (int j = 0; j < 2; ++j) {
                acc[i][j] = __builtin_amdgcn_wmma_f32_16x16x32_bf16(
                    false, a[i], false, b[j], (short)0, acc[i][j], false, false);
            }
        }
    }

    // C/D layout: element e -> row (tile_m + e + 8*half), col (tile_n + r)
#pragma unroll
    for (int i = 0; i < 4; ++i) {
#pragma unroll
        for (int e = 0; e < 8; ++e) {
            const int row = m0 + i * 16 + e + 8 * half;
            const int col = n0 + r;
            size_t o0 = (size_t)row * N + col;
            size_t o1 = o0 + 16;
            float v0 = acc[i][0][e], v1 = acc[i][1][e];
            if (resid) { v0 += resid[o0]; v1 += resid[o1]; }
            C[o0] = v0; C[o1] = v1;
        }
    }
}

// -----------------------------------------------------------------------------
// Depthwise causal conv (width 4) + bias + SiLU over the xBC slice of zxbcdt.
// -----------------------------------------------------------------------------
__global__ __launch_bounds__(256)
void conv_silu_kernel(const float* __restrict__ zxbcdt,
                      const float* __restrict__ conv_w,
                      const float* __restrict__ conv_b,
                      float* __restrict__ xbc_out) {
    int idx = blockIdx.x * blockDim.x + threadIdx.x;
    const int total = MROWS * CONV_DIM;
    if (idx >= total) return;
    const int c  = idx % CONV_DIM;
    const int t  = (idx / CONV_DIM) % SEQLEN;
    const int bb = idx / (CONV_DIM * SEQLEN);

    const float* base = zxbcdt + (size_t)bb * SEQLEN * D_IN_PROJ + D_INNER + c;
    float s = conv_b[c];
#pragma unroll
    for (int k = 0; k < D_CONV; ++k) {
        const int tt = t + k - (D_CONV - 1);
        const float xv = (tt >= 0) ? base[(size_t)tt * D_IN_PROJ] : 0.f;
        s += xv * conv_w[c * D_CONV + k];
    }
    xbc_out[idx] = s / (1.f + __expf(-s));   // SiLU
}

// -----------------------------------------------------------------------------
// dt = softplus(dt_raw + dt_bias);  dA = exp(dt * (-exp(A_log)))
// -----------------------------------------------------------------------------
__global__ __launch_bounds__(256)
void dt_kernel(const float* __restrict__ zxbcdt,
               const float* __restrict__ dt_bias,
               const float* __restrict__ A_log,
               float* __restrict__ dtb,
               float* __restrict__ dAe) {
    int idx = blockIdx.x * blockDim.x + threadIdx.x;
    const int total = MROWS * NHEADS;
    if (idx >= total) return;
    const int h   = idx % NHEADS;
    const int row = idx / NHEADS;
    const float d  = zxbcdt[(size_t)row * D_IN_PROJ + D_INNER + CONV_DIM + h] + dt_bias[h];
    const float sp = (d > 20.f) ? d : log1pf(__expf(d));
    const float Ah = -__expf(A_log[h]);
    dtb[idx] = sp;
    dAe[idx] = __expf(sp * Ah);
}

// -----------------------------------------------------------------------------
// Selective scan.  p (headdim) is independent in the recurrence -> split each
// (batch, head) into 4 p-chunks of 16 -> 256 blocks.  256 threads:
// thread = p_local*16 + ng; each owns 8 state columns in registers.
// B_t/C_t staged in LDS per step via GLOBAL_LOAD_ASYNC_TO_LDS (ASYNCcnt)
// when available; n-reduction via 4 wave32 shuffles in the 16-lane group.
// -----------------------------------------------------------------------------
__global__ __launch_bounds__(256)
void scan_kernel(const float* __restrict__ xbc,   // (B,L,CONV_DIM): [xs | B | C]
                 const float* __restrict__ dtb,   // (B,L,H)
                 const float* __restrict__ dAe,   // (B,L,H)
                 const float* __restrict__ Dp,    // (H)
                 float* __restrict__ y) {         // (B,L,D_INNER)
    const int bh = blockIdx.x >> 2;       // (batch, head)
    const int pc = blockIdx.x & 3;        // p chunk: p = pc*16 + p_local
    const int bb = bh / NHEADS;
    const int h  = bh % NHEADS;
    const int tid = threadIdx.x;
    const int p_local = tid >> 4;         // 0..15
    const int ng      = tid & 15;         // 0..15
    const int n0      = ng * 8;
    const int p       = pc * 16 + p_local;

    __shared__ float sBC[2 * D_STATE];    // [B(128) | C(128)]
    __shared__ float sX[16];              // xs chunk for this block

    float S[8];
#pragma unroll
    for (int j = 0; j < 8; ++j) S[j] = 0.f;

    const float Dh = Dp[h];
    const float* xrow  = xbc + (size_t)bb * SEQLEN * CONV_DIM;
    const float* dtrow = dtb + (size_t)bb * SEQLEN * NHEADS + h;
    const float* darow = dAe + (size_t)bb * SEQLEN * NHEADS + h;
    float* yrow = y + (size_t)bb * SEQLEN * D_INNER + h * HEADDIM;

    for (int t = 0; t < SEQLEN; ++t) {
        const float* xb = xrow + (size_t)t * CONV_DIM;
        // tid 0..127 stages B, 128..255 stages C (C starts at D_INNER+128)
#if defined(HAVE_ASYNC_LDS)
        __builtin_amdgcn_global_load_async_to_lds_b32(
            (gptr_i32)(void*)(xb + D_INNER + tid),
            (lptr_i32)(void*)&sBC[tid],
            /*offset=*/0, /*cpol=*/0);
        if (tid < 16) sX[tid] = xb[h * HEADDIM + pc * 16 + tid];
        asm volatile("s_wait_asynccnt 0" ::: "memory");
#else
        sBC[tid] = xb[D_INNER + tid];
        if (tid < 16) sX[tid] = xb[h * HEADDIM + pc * 16 + tid];
#endif
        __syncthreads();

        const float dA_t = darow[(size_t)t * NHEADS];
        const float dt_t = dtrow[(size_t)t * NHEADS];
        const float xv   = sX[p_local];
        const float xdt  = xv * dt_t;

        float yp = 0.f;
#pragma unroll
        for (int j = 0; j < 8; ++j) {
            const float s = dA_t * S[j] + xdt * sBC[n0 + j];
            S[j] = s;
            yp += s * sBC[D_STATE + n0 + j];
        }
        // reduce across the 16 consecutive lanes sharing p
        yp += __shfl_xor(yp, 1, 32);
        yp += __shfl_xor(yp, 2, 32);
        yp += __shfl_xor(yp, 4, 32);
        yp += __shfl_xor(yp, 8, 32);
        if (ng == 0) yrow[(size_t)t * D_INNER + p] = yp + Dh * xv;
        __syncthreads();
    }
}

// -----------------------------------------------------------------------------
// y = y * silu(z); yn = y * rsqrt(mean(y^2)+1e-5) * norm_w, emitted in bf16
// (feeds GEMM2 directly).  One block per row; 8 consecutive channels/thread.
// -----------------------------------------------------------------------------
__global__ __launch_bounds__(256)
void gate_norm_kernel(const float* __restrict__ y,
                      const float* __restrict__ zxbcdt,  // z = first D_INNER
                      const float* __restrict__ norm_w,
                      __bf16* __restrict__ yn) {
    const int row = blockIdx.x;
    const float* yr = y + (size_t)row * D_INNER;
    const float* zr = zxbcdt + (size_t)row * D_IN_PROJ;
    __bf16* outr = yn + (size_t)row * D_INNER;

    __shared__ float red[8];
    const int c0 = threadIdx.x * 8;
    float g[8];
    float ss = 0.f;
#pragma unroll
    for (int i = 0; i < 8; ++i) {
        const float z  = zr[c0 + i];
        const float gv = yr[c0 + i] * (z / (1.f + __expf(-z)));
        g[i] = gv;
        ss += gv * gv;
    }
#pragma unroll
    for (int o = 16; o > 0; o >>= 1) ss += __shfl_xor(ss, o, 32);
    const int wave = threadIdx.x >> 5, lane = threadIdx.x & 31;
    if (lane == 0) red[wave] = ss;
    __syncthreads();
    if (wave == 0) {
        float v = (lane < 8) ? red[lane] : 0.f;
#pragma unroll
        for (int o = 4; o > 0; o >>= 1) v += __shfl_xor(v, o, 32);
        if (lane == 0) red[0] = v;
    }
    __syncthreads();
    const float rms = rsqrtf(red[0] / (float)D_INNER + 1e-5f);
    v8bf outv;
#pragma unroll
    for (int i = 0; i < 8; ++i) outv[i] = (__bf16)(g[i] * rms * norm_w[c0 + i]);
    *(v8bf*)(outr + c0) = outv;
}

// -----------------------------------------------------------------------------
extern "C" void kernel_launch(void* const* d_in, const int* in_sizes, int n_in,
                              void* d_out, int out_size, void* d_ws, size_t ws_size,
                              hipStream_t stream) {
    (void)in_sizes; (void)n_in; (void)out_size; (void)ws_size;
    const float* x          = (const float*)d_in[0];
    const float* in_proj_w  = (const float*)d_in[1];
    const float* conv_w     = (const float*)d_in[2];
    const float* conv_b     = (const float*)d_in[3];
    const float* dt_bias    = (const float*)d_in[4];
    const float* A_log      = (const float*)d_in[5];
    const float* Dp         = (const float*)d_in[6];
    const float* norm_w     = (const float*)d_in[7];
    const float* out_proj_w = (const float*)d_in[8];
    float* out = (float*)d_out;

    char* ws = (char*)d_ws;
    size_t off = 0;
    auto alloc = [&](size_t bytes) {
        void* p = (void*)(ws + off);
        off += (bytes + 255) & ~(size_t)255;
        return p;
    };
    float*  zxbcdt  = (float*) alloc((size_t)MROWS * D_IN_PROJ * 4);  // 71.8 MB
    float*  xbc     = (float*) alloc((size_t)MROWS * CONV_DIM * 4);   // 37.7 MB
    float*  dtb     = (float*) alloc((size_t)MROWS * NHEADS * 4);
    float*  dAe     = (float*) alloc((size_t)MROWS * NHEADS * 4);
    float*  y       = (float*) alloc((size_t)MROWS * D_INNER * 4);    // 33.5 MB
    __bf16* x_bf    = (__bf16*)alloc((size_t)MROWS * D_MODEL * 2);    //  8.4 MB
    __bf16* w1_bf   = (__bf16*)alloc((size_t)D_IN_PROJ * D_MODEL * 2);//  9.0 MB
    __bf16* w2_bf   = (__bf16*)alloc((size_t)D_MODEL * D_INNER * 2);  //  4.2 MB
    __bf16* yn_bf   = (__bf16*)alloc((size_t)MROWS * D_INNER * 2);    // 16.8 MB

    // 0) one-pass f32 -> bf16 for GEMM operands
    {
        int n8 = (MROWS * D_MODEL) / 8;
        f32_to_bf16_kernel<<<(n8 + 255) / 256, 256, 0, stream>>>(x, (v8bf*)x_bf, n8);
        n8 = (D_IN_PROJ * D_MODEL) / 8;
        f32_to_bf16_kernel<<<(n8 + 255) / 256, 256, 0, stream>>>(in_proj_w, (v8bf*)w1_bf, n8);
        n8 = (D_MODEL * D_INNER) / 8;
        f32_to_bf16_kernel<<<(n8 + 255) / 256, 256, 0, stream>>>(out_proj_w, (v8bf*)w2_bf, n8);
    }
    // 1) in_proj GEMM: zxbcdt[4096,4384] = x[4096,1024] @ in_proj_w^T
    {
        dim3 grid(D_IN_PROJ / 32, MROWS / 256);
        wmma_gemm_nt<<<grid, 128, 0, stream>>>(x_bf, w1_bf, nullptr, zxbcdt,
                                               MROWS, D_IN_PROJ, D_MODEL);
    }
    // 2) causal conv + SiLU
    {
        const int total = MROWS * CONV_DIM;
        conv_silu_kernel<<<(total + 255) / 256, 256, 0, stream>>>(zxbcdt, conv_w,
                                                                  conv_b, xbc);
    }
    // 3) dt softplus + dA
    {
        const int total = MROWS * NHEADS;
        dt_kernel<<<(total + 255) / 256, 256, 0, stream>>>(zxbcdt, dt_bias, A_log,
                                                           dtb, dAe);
    }
    // 4) selective scan (sequential over L; parallel over b*h*4 p-chunks)
    scan_kernel<<<BATCH * NHEADS * 4, 256, 0, stream>>>(xbc, dtb, dAe, Dp, y);

    // 5) gating + RMSNorm, emitted as bf16 GEMM2 operand
    gate_norm_kernel<<<MROWS, 256, 0, stream>>>(y, zxbcdt, norm_w, yn_bf);

    // 6) out_proj GEMM + residual: out = x + yn @ out_proj_w^T
    {
        dim3 grid(D_MODEL / 32, MROWS / 256);
        wmma_gemm_nt<<<grid, 128, 0, stream>>>(yn_bf, w2_bf, x, out,
                                               MROWS, D_MODEL, D_INNER);
    }
}
